// GCNTox21_19808389169323
// MI455X (gfx1250) — matrix operand order repaired and checked by reading, not compiled, not run on hardware
//
#include <hip/hip_runtime.h>

// ---------------------------------------------------------------------------
// Types for WMMA fragments (gfx1250, wave32)
// ---------------------------------------------------------------------------
typedef __attribute__((ext_vector_type(16))) __bf16 v16bf;
typedef __attribute__((ext_vector_type(8)))  __bf16 v8bf;
typedef __attribute__((ext_vector_type(8)))  float  v8f;

#define BN_EPS 1e-5f

// ---------------------------------------------------------------------------
// Pack a row-major fp32 weight [K, Nout] into bf16 B-fragment order.
// Packed index = ((nt*nKT + kt)*32 + lane)*16 + e
//   column n = nt*16 + (lane & 15)
//   k        = kt*32 + (e&8)*2 + (lane>>4)*8 + (e&7)     (B frag: table order, matches A)
// ---------------------------------------------------------------------------
__global__ void pack_b_kernel(const float* __restrict__ W, __bf16* __restrict__ packed,
                              int K, int Nout) {
    int tid = blockIdx.x * blockDim.x + threadIdx.x;
    int total = K * Nout;
    if (tid >= total) return;
    int nKT = K >> 5;
    int e    = tid & 15;
    int lane = (tid >> 4) & 31;
    int t    = tid >> 9;
    int kt   = t % nKT;
    int nt   = t / nKT;
    int n = nt * 16 + (lane & 15);
    int k = kt * 32 + (e & 8) * 2 + (lane >> 4) * 8 + (e & 7);
    packed[tid] = (__bf16)W[k * Nout + n];
}

// ---------------------------------------------------------------------------
// Node embedding: h = x @ Wn + b  -> bf16 [N, 64]
// ---------------------------------------------------------------------------
__global__ void embed_kernel(const float* __restrict__ x, const float* __restrict__ Wn,
                             const float* __restrict__ bn, __bf16* __restrict__ hbf,
                             int nNodes) {
    int tid = blockIdx.x * blockDim.x + threadIdx.x;
    if (tid >= nNodes * 64) return;
    int n = tid >> 6;
    int f = tid & 63;
    const float* xr = x + n * 32;
    float s = bn[f];
#pragma unroll
    for (int k = 0; k < 32; ++k) s = fmaf(xr[k], Wn[k * 64 + f], s);
    hbf[tid] = (__bf16)s;
}

// ---------------------------------------------------------------------------
// Degree / graph-count precompute
// ---------------------------------------------------------------------------
__global__ void deg_kernel(const int* __restrict__ dst, int* __restrict__ cnt, int nE) {
    int tid = blockIdx.x * blockDim.x + threadIdx.x;
    if (tid < nE) atomicAdd(&cnt[dst[tid]], 1);
}

__global__ void node_stats_kernel(const int* __restrict__ cnt, float* __restrict__ inv_deg,
                                  const int* __restrict__ batch, float* __restrict__ gcnt,
                                  int nN) {
    int tid = blockIdx.x * blockDim.x + threadIdx.x;
    if (tid >= nN) return;
    int c = cnt[tid];
    inv_deg[tid] = 1.0f / (float)(c > 1 ? c : 1);
    atomicAdd(&gcnt[batch[tid]], 1.0f);
}

// ---------------------------------------------------------------------------
// Gather one 16-edge A tile for z = [x_i, x_j - x_i] into WMMA A fragments.
// A layout (16-bit, 16x32): lane L -> row L%16; element e -> k =
//   (e<8?0:16) + 8*(L>=16) + (e&7)
// ---------------------------------------------------------------------------
template <int D, int NKT>
__device__ __forceinline__ void gather_a(const __bf16* __restrict__ hbf,
                                         int sn, int dn, int hl,
                                         v16bf (&afr)[NKT]) {
    const __bf16* xi = hbf + (size_t)dn * D;
    const __bf16* xj = hbf + (size_t)sn * D;
#pragma unroll
    for (int kt = 0; kt < NKT; ++kt) {
        const int base = kt * 32;
        v8bf lo, hi;
        if (base < D) {  // x_i region
            const int fo = base + hl * 8;
            lo = *(const v8bf*)(xi + fo);
            hi = *(const v8bf*)(xi + fo + 16);
        } else {         // x_j - x_i region
            const int fo = (base - D) + hl * 8;
            v8bf li = *(const v8bf*)(xi + fo);
            v8bf hi_i = *(const v8bf*)(xi + fo + 16);
            v8bf lj = *(const v8bf*)(xj + fo);
            v8bf hj = *(const v8bf*)(xj + fo + 16);
#pragma unroll
            for (int j = 0; j < 8; ++j) {
                lo[j] = (__bf16)((float)lj[j] - (float)li[j]);
                hi[j] = (__bf16)((float)hj[j] - (float)hi_i[j]);
            }
        }
        v16bf a;
#pragma unroll
        for (int j = 0; j < 8; ++j) { a[j] = lo[j]; a[8 + j] = hi[j]; }
        afr[kt] = a;
    }
}

// ---------------------------------------------------------------------------
// Fused EdgeConv message MLP over a 32-edge tile (one wave32 per block).
// Two 16-row A tiles share every loaded B fragment (halves weight traffic).
//   z = [x_i, x_j - x_i]  (DIN = 2*D)
//   hidden = relu(z @ Wa + ba)   (H)
//   msg    = hidden @ Wb + bb    (DOUT), scattered with atomics to accum[dst]
// All matmul work via v_wmma_f32_16x16x32_bf16.
// ---------------------------------------------------------------------------
template <int D, int H, int DOUT>
__global__ void __launch_bounds__(32)
edgeconv_kernel(const __bf16* __restrict__ hbf,
                const int* __restrict__ srcIdx, const int* __restrict__ dstIdx,
                const __bf16* __restrict__ pWa, const float* __restrict__ ba,
                const __bf16* __restrict__ pWb, const float* __restrict__ bb,
                float* __restrict__ accum, int nEdges) {
    constexpr int DIN  = 2 * D;
    constexpr int NKT1 = DIN / 32;   // k-tiles for GEMM1
    constexpr int NNT1 = H / 16;     // n-tiles for GEMM1
    constexpr int NKT2 = H / 32;     // k-tiles for GEMM2
    constexpr int NNT2 = DOUT / 16;  // n-tiles for GEMM2

    __shared__ __align__(16) __bf16 hid[32 * H];
    __shared__ int sdst[32];

    const int lane = threadIdx.x & 31;
    const int mrow = lane & 15;   // matrix row within sub-tile / output column
    const int hl   = lane >> 4;   // lane half

    const long long tileBase = (long long)blockIdx.x * 32;

    // sub-tile 0: edges tileBase + mrow ; sub-tile 1: edges tileBase + 16 + mrow
    const int e0 = (int)tileBase + mrow;
    const int e1 = e0 + 16;
    const bool v0 = (e0 < nEdges);
    const bool v1 = (e1 < nEdges);
    const int ec0 = v0 ? e0 : 0;
    const int ec1 = v1 ? e1 : 0;
    const int dn0 = dstIdx[ec0];
    const int sn0 = srcIdx[ec0];
    const int dn1 = dstIdx[ec1];
    const int sn1 = srcIdx[ec1];
    if (hl == 0) {
        sdst[mrow]      = v0 ? dn0 : -1;
        sdst[16 + mrow] = v1 ? dn1 : -1;
    }

    // ---- gather A fragments for both sub-tiles --------------------------
    v16bf afr0[NKT1], afr1[NKT1];
    gather_a<D, NKT1>(hbf, sn0, dn0, hl, afr0);
    gather_a<D, NKT1>(hbf, sn1, dn1, hl, afr1);

    // ---- GEMM1: hidden = relu(z @ Wa + ba) -> LDS (bf16) ----------------
#pragma unroll
    for (int nt = 0; nt < NNT1; ++nt) {
        v8f acc0 = {}, acc1 = {};
#pragma unroll
        for (int kt = 0; kt < NKT1; ++kt) {
            v16bf b = *(const v16bf*)(pWa + (((size_t)nt * NKT1 + kt) * 32 + lane) * 16);
            acc0 = __builtin_amdgcn_wmma_f32_16x16x32_bf16(
                false, afr0[kt], false, b, (short)0, acc0, false, false);
            acc1 = __builtin_amdgcn_wmma_f32_16x16x32_bf16(
                false, afr1[kt], false, b, (short)0, acc1, false, false);
        }
        const float bias = ba[nt * 16 + mrow];
#pragma unroll
        for (int r = 0; r < 8; ++r) {
            float va = acc0[r] + bias;
            float vb = acc1[r] + bias;
            va = va > 0.0f ? va : 0.0f;
            vb = vb > 0.0f ? vb : 0.0f;
            hid[(r + 8 * hl) * H + nt * 16 + mrow]        = (__bf16)va;
            hid[(16 + r + 8 * hl) * H + nt * 16 + mrow]   = (__bf16)vb;
        }
    }
    __syncthreads();

    // ---- reload hidden tiles as A fragments (transpose via LDS) ---------
    v16bf a20[NKT2], a21[NKT2];
#pragma unroll
    for (int kt = 0; kt < NKT2; ++kt) {
        const int off0 = mrow * H + kt * 32 + hl * 8;
        const int off1 = (16 + mrow) * H + kt * 32 + hl * 8;
        v8bf lo0 = *(const v8bf*)(hid + off0);
        v8bf hi0 = *(const v8bf*)(hid + off0 + 16);
        v8bf lo1 = *(const v8bf*)(hid + off1);
        v8bf hi1 = *(const v8bf*)(hid + off1 + 16);
        v16bf a, b;
#pragma unroll
        for (int j = 0; j < 8; ++j) {
            a[j] = lo0[j]; a[8 + j] = hi0[j];
            b[j] = lo1[j]; b[8 + j] = hi1[j];
        }
        a20[kt] = a;
        a21[kt] = b;
    }

    // ---- GEMM2: msg = hidden @ Wb + bb, scatter-add at dst --------------
#pragma unroll
    for (int nt = 0; nt < NNT2; ++nt) {
        v8f acc0 = {}, acc1 = {};
#pragma unroll
        for (int kt = 0; kt < NKT2; ++kt) {
            v16bf b = *(const v16bf*)(pWb + (((size_t)nt * NKT2 + kt) * 32 + lane) * 16);
            acc0 = __builtin_amdgcn_wmma_f32_16x16x32_bf16(
                false, a20[kt], false, b, (short)0, acc0, false, false);
            acc1 = __builtin_amdgcn_wmma_f32_16x16x32_bf16(
                false, a21[kt], false, b, (short)0, acc1, false, false);
        }
        const float bias = bb[nt * 16 + mrow];
#pragma unroll
        for (int r = 0; r < 8; ++r) {
            const int node0 = sdst[r + 8 * hl];
            const int node1 = sdst[16 + r + 8 * hl];
            if (node0 >= 0)
                atomicAdd(&accum[(size_t)node0 * DOUT + nt * 16 + mrow], acc0[r] + bias);
            if (node1 >= 0)
                atomicAdd(&accum[(size_t)node1 * DOUT + nt * 16 + mrow], acc1[r] + bias);
        }
    }
}

// ---------------------------------------------------------------------------
// Node finalize: h = relu(BN(accum * inv_deg)) -> bf16 for the next gather
// ---------------------------------------------------------------------------
template <int DOUT>
__global__ void finalize_kernel(const float* __restrict__ accum,
                                const float* __restrict__ inv_deg,
                                const float* __restrict__ g, const float* __restrict__ beta,
                                __bf16* __restrict__ hbf, int nN) {
    int tid = blockIdx.x * blockDim.x + threadIdx.x;
    if (tid >= nN * DOUT) return;
    const int n = tid / DOUT;
    const int f = tid % DOUT;
    const float scale = g[f] * rsqrtf(1.0f + BN_EPS);
    float v = accum[tid] * inv_deg[n] * scale + beta[f];
    v = v > 0.0f ? v : 0.0f;
    hbf[tid] = (__bf16)v;
}

// ---------------------------------------------------------------------------
// Global mean pool (sum stage) + classifier head
// ---------------------------------------------------------------------------
__global__ void pool_kernel(const __bf16* __restrict__ hbf, const int* __restrict__ batch,
                            float* __restrict__ pooled, int nN) {
    int tid = blockIdx.x * blockDim.x + threadIdx.x;
    if (tid >= nN * 32) return;
    const int n = tid >> 5;
    const int f = tid & 31;
    atomicAdd(&pooled[batch[n] * 32 + f], (float)hbf[tid]);
}

__global__ void head_kernel(const float* __restrict__ pooled, const float* __restrict__ gcnt,
                            const float* __restrict__ Wfc, const float* __restrict__ bfc,
                            float* __restrict__ out, int nG) {
    int gidx = blockIdx.x * blockDim.x + threadIdx.x;
    if (gidx >= nG) return;
    const float inv = 1.0f / fmaxf(gcnt[gidx], 1.0f);
    float p[32];
#pragma unroll
    for (int k = 0; k < 32; ++k) p[k] = pooled[gidx * 32 + k] * inv;
#pragma unroll
    for (int o = 0; o < 5; ++o) {
        float s = bfc[o];
#pragma unroll
        for (int k = 0; k < 32; ++k) s = fmaf(p[k], Wfc[k * 5 + o], s);
        out[gidx * 5 + o] = 1.0f / (1.0f + expf(-s));
    }
}

// ---------------------------------------------------------------------------
// Host launcher
// ---------------------------------------------------------------------------
extern "C" void kernel_launch(void* const* d_in, const int* in_sizes, int n_in,
                              void* d_out, int out_size, void* d_ws, size_t ws_size,
                              hipStream_t stream) {
    const int N = 50000, E = 600000, G = 2048;

    const float* x     = (const float*)d_in[0];
    // d_in[1..3]: edge_attr / We / be — unused by the reference output
    const float* Wn    = (const float*)d_in[4];
    const float* bnode = (const float*)d_in[5];
    const float* W1a = (const float*)d_in[6];   const float* b1a = (const float*)d_in[7];
    const float* W1b = (const float*)d_in[8];   const float* b1b = (const float*)d_in[9];
    const float* W2a = (const float*)d_in[10];  const float* b2a = (const float*)d_in[11];
    const float* W2b = (const float*)d_in[12];  const float* b2b = (const float*)d_in[13];
    const float* W3a = (const float*)d_in[14];  const float* b3a = (const float*)d_in[15];
    const float* W3b = (const float*)d_in[16];  const float* b3b = (const float*)d_in[17];
    const float* g1 = (const float*)d_in[18];   const float* beta1 = (const float*)d_in[19];
    const float* g2 = (const float*)d_in[20];   const float* beta2 = (const float*)d_in[21];
    const float* g3 = (const float*)d_in[22];   const float* beta3 = (const float*)d_in[23];
    const float* Wfc = (const float*)d_in[24];  const float* bfc = (const float*)d_in[25];
    const int* edge_index = (const int*)d_in[26];
    const int* batch      = (const int*)d_in[27];
    float* out = (float*)d_out;

    const int* srcIdx = edge_index;       // edge_index[0] = j (source)
    const int* dstIdx = edge_index + E;   // edge_index[1] = i (target / aggregation)

    // ---- workspace carving ---------------------------------------------
    char* wsb = (char*)d_ws;
    size_t off = 0;
    auto carve = [&](size_t bytes) -> void* {
        void* p = wsb + off;
        off = (off + bytes + 255) & ~(size_t)255;
        return p;
    };
    __bf16* hbf   = (__bf16*)carve((size_t)N * 128 * sizeof(__bf16));
    float* accum  = (float*)carve((size_t)N * 128 * sizeof(float));
    float* invdeg = (float*)carve((size_t)N * sizeof(float));
    int*   cnt    = (int*)carve((size_t)N * sizeof(int));
    float* gcnt   = (float*)carve((size_t)G * sizeof(float));
    float* pooled = (float*)carve((size_t)G * 32 * sizeof(float));
    __bf16* p1a = (__bf16*)carve((size_t)128 * 128 * 2);
    __bf16* p1b = (__bf16*)carve((size_t)128 * 128 * 2);
    __bf16* p2a = (__bf16*)carve((size_t)256 * 64 * 2);
    __bf16* p2b = (__bf16*)carve((size_t)64 * 64 * 2);
    __bf16* p3a = (__bf16*)carve((size_t)128 * 32 * 2);
    __bf16* p3b = (__bf16*)carve((size_t)32 * 32 * 2);

    // ---- init / precompute ---------------------------------------------
    hipMemsetAsync(cnt, 0, (size_t)N * sizeof(int), stream);
    hipMemsetAsync(gcnt, 0, (size_t)G * sizeof(float), stream);
    hipMemsetAsync(pooled, 0, (size_t)G * 32 * sizeof(float), stream);

    auto packGrid = [](int K, int Nc) { return (K * Nc + 255) / 256; };
    pack_b_kernel<<<packGrid(128, 128), 256, 0, stream>>>(W1a, p1a, 128, 128);
    pack_b_kernel<<<packGrid(128, 128), 256, 0, stream>>>(W1b, p1b, 128, 128);
    pack_b_kernel<<<packGrid(256, 64),  256, 0, stream>>>(W2a, p2a, 256, 64);
    pack_b_kernel<<<packGrid(64, 64),   256, 0, stream>>>(W2b, p2b, 64, 64);
    pack_b_kernel<<<packGrid(128, 32),  256, 0, stream>>>(W3a, p3a, 128, 32);
    pack_b_kernel<<<packGrid(32, 32),   256, 0, stream>>>(W3b, p3b, 32, 32);

    deg_kernel<<<(E + 255) / 256, 256, 0, stream>>>(dstIdx, cnt, E);
    node_stats_kernel<<<(N + 255) / 256, 256, 0, stream>>>(cnt, invdeg, batch, gcnt, N);

    embed_kernel<<<(N * 64 + 255) / 256, 256, 0, stream>>>(x, Wn, bnode, hbf, N);

    const int nTiles = (E + 31) / 32;   // 32 edges per wave/block

    // ---- layer 1: D=64 -> H=128 -> DOUT=128 ----------------------------
    hipMemsetAsync(accum, 0, (size_t)N * 128 * sizeof(float), stream);
    edgeconv_kernel<64, 128, 128><<<nTiles, 32, 0, stream>>>(
        hbf, srcIdx, dstIdx, p1a, b1a, p1b, b1b, accum, E);
    finalize_kernel<128><<<(N * 128 + 255) / 256, 256, 0, stream>>>(
        accum, invdeg, g1, beta1, hbf, N);

    // ---- layer 2: D=128 -> H=64 -> DOUT=64 -----------------------------
    hipMemsetAsync(accum, 0, (size_t)N * 64 * sizeof(float), stream);
    edgeconv_kernel<128, 64, 64><<<nTiles, 32, 0, stream>>>(
        hbf, srcIdx, dstIdx, p2a, b2a, p2b, b2b, accum, E);
    finalize_kernel<64><<<(N * 64 + 255) / 256, 256, 0, stream>>>(
        accum, invdeg, g2, beta2, hbf, N);

    // ---- layer 3: D=64 -> H=32 -> DOUT=32 ------------------------------
    hipMemsetAsync(accum, 0, (size_t)N * 32 * sizeof(float), stream);
    edgeconv_kernel<64, 32, 32><<<nTiles, 32, 0, stream>>>(
        hbf, srcIdx, dstIdx, p3a, b3a, p3b, b3b, accum, E);
    finalize_kernel<32><<<(N * 32 + 255) / 256, 256, 0, stream>>>(
        accum, invdeg, g3, beta3, hbf, N);

    // ---- pool + head ----------------------------------------------------
    pool_kernel<<<(N * 32 + 255) / 256, 256, 0, stream>>>(hbf, batch, pooled, N);
    head_kernel<<<(G + 127) / 128, 128, 0, stream>>>(pooled, gcnt, Wfc, bfc, out, G);
}
